// MultiHeadAttention_66073776882364
// MI455X (gfx1250) — matrix operand order repaired and checked by reading, not compiled, not run on hardware
//
#include <hip/hip_runtime.h>
#include <hip/hip_bf16.h>

typedef __attribute__((ext_vector_type(16))) _Float16 v16h;
typedef __attribute__((ext_vector_type(8)))  _Float16 v8h;
typedef __attribute__((ext_vector_type(8)))  float    v8f;

#define NBATCH 4
#define T_SEQ  2048
#define CDIM   1024
#define NHEAD  16
#define DHEAD  64
#define BHTOT  (NBATCH * NHEAD)

// fold 1/sqrt(64) * log2(e) into Q so softmax = exp2(s - m)
#define QSCALE 0.18033688011112042f

__device__ inline v16h pack16(v8h lo, v8h hi) {
    union { v16h v; v8h h[2]; } u;
    u.h[0] = lo; u.h[1] = hi;
    return u.v;
}

// 16 contiguous halves (32B-aligned)
__device__ inline v16h load16h(const _Float16* p) {
    const v8h* q = (const v8h*)p;
    return pack16(q[0], q[1]);
}

// A-operand from a row-major f16 row chunk base (K-chunk already applied):
// halves 0..15 at +abase contiguous  (abase = lane<16 ? 0 : 16)
__device__ inline v16h loadA_h(const _Float16* rowc, int abase) {
    return pack16(*(const v8h*)(rowc + abase), *(const v8h*)(rowc + 8 + abase));
}

// 8 contiguous f32 -> 8 halves
__device__ inline v8h cvt8(const float* p) {
    const float4* f = (const float4*)p;
    float4 x0 = f[0], x1 = f[1];
    v8h r;
    r[0] = (_Float16)x0.x; r[1] = (_Float16)x0.y; r[2] = (_Float16)x0.z; r[3] = (_Float16)x0.w;
    r[4] = (_Float16)x1.x; r[5] = (_Float16)x1.y; r[6] = (_Float16)x1.z; r[7] = (_Float16)x1.w;
    return r;
}

__device__ inline v16h loadA_f32(const float* rowc, int abase) {
    return pack16(cvt8(rowc + abase), cvt8(rowc + 8 + abase));
}

// B-operand from 16 contiguous f32 (koff already applied)
__device__ inline v16h loadB_f32(const float* p) {
    return pack16(cvt8(p), cvt8(p + 8));
}

__device__ inline v8f wmma16(v16h a, v16h b, v8f c) {
    return __builtin_amdgcn_wmma_f32_16x16x32_f16(false, a, false, b, (short)0, c, false, false);
}

// ---------------------------------------------------------------------------
// Kernel 1: fused Q/K/V projections.
//   Qh[bh][t][e] (f16, pre-scaled by log2e/8), Kh[bh][t][e] (f16),
//   Vt[bh][e][t] (f16, transposed for PV B-operand loads).
// One wave = 16 rows of one (b,h). 8 waves/block, 1024 blocks.
// ---------------------------------------------------------------------------
__global__ __launch_bounds__(256) void qkv_proj_kernel(
    const float* __restrict__ x,
    const float* __restrict__ Wq, const float* __restrict__ bq,
    const float* __restrict__ Wk, const float* __restrict__ bk,
    const float* __restrict__ Wv, const float* __restrict__ bv,
    _Float16* __restrict__ Qh, _Float16* __restrict__ Kh, _Float16* __restrict__ Vt)
{
    const int lane  = threadIdx.x & 31;
    const int wave  = threadIdx.x >> 5;
    const int w     = blockIdx.x * 8 + wave;
    const int bh    = w >> 7;            // 0..63
    const int t0    = (w & 127) << 4;    // 0..2032
    const int b     = bh >> 4;
    const int h     = bh & 15;

    const int mrow  = lane & 15;                  // A-operand row
    const int abase = (lane < 16) ? 0 : 16;       // A-operand K group base
    const int koff  = (lane < 16) ? 0 : 16;       // B-operand K half
    const int off   = (lane < 16) ? 0 : 8;        // C/D row offset
    const int ncol  = lane & 15;                  // C/D / B column

    // Load A = x tile rows (f32 -> f16), K chunks c=0,1
    v16h a[2];
    {
        const float* xrow = x + ((size_t)(b * T_SEQ + t0 + mrow)) * CDIM + h * DHEAD;
        #pragma unroll
        for (int c = 0; c < 2; ++c)
            a[c] = loadA_f32(xrow + c * 32, abase);
    }

    #pragma unroll
    for (int proj = 0; proj < 3; ++proj) {
        const float* W    = (proj == 0) ? Wq : (proj == 1) ? Wk : Wv;
        const float* bias = (proj == 0) ? bq : (proj == 1) ? bk : bv;

        v8f acc[4];
        #pragma unroll
        for (int nt = 0; nt < 4; ++nt) {
            v8f c0 = {0.f, 0.f, 0.f, 0.f, 0.f, 0.f, 0.f, 0.f};
            #pragma unroll
            for (int c = 0; c < 2; ++c) {
                v16h bm = loadB_f32(W + (size_t)(nt * 16 + ncol) * 64 + c * 32 + koff);
                c0 = wmma16(a[c], bm, c0);
            }
            acc[nt] = c0;
        }

        #pragma unroll
        for (int nt = 0; nt < 4; ++nt) {
            const int e = nt * 16 + ncol;
            const float bb = bias[e];
            if (proj == 0) {
                #pragma unroll
                for (int r = 0; r < 8; ++r)
                    Qh[((size_t)bh * T_SEQ + t0 + r + off) * DHEAD + e] =
                        (_Float16)((acc[nt][r] + bb) * QSCALE);
            } else if (proj == 1) {
                #pragma unroll
                for (int r = 0; r < 8; ++r)
                    Kh[((size_t)bh * T_SEQ + t0 + r + off) * DHEAD + e] =
                        (_Float16)(acc[nt][r] + bb);
            } else {
                v8h tmp;
                #pragma unroll
                for (int r = 0; r < 8; ++r)
                    tmp[r] = (_Float16)(acc[nt][r] + bb);
                // rows r map to consecutive t -> contiguous 16B store in Vt
                *(v8h*)(Vt + (size_t)bh * DHEAD * T_SEQ + (size_t)e * T_SEQ + t0 + off) = tmp;
            }
        }
    }
}

// ---------------------------------------------------------------------------
// Kernel 2: flash attention. One wave = 16 query rows of one (b,h).
// Loop over 64-key blocks: 8 WMMAs S=Q*K^T, online softmax, 8 WMMAs O+=P*V.
// P routed via private per-wave LDS tile (C/D layout -> A layout).
// Next K/V block prefetched into WGP$ (global_prefetch_b8) while computing.
// ---------------------------------------------------------------------------
__global__ __launch_bounds__(256) void flash_attn_kernel(
    const _Float16* __restrict__ Qh, const _Float16* __restrict__ Kh,
    const _Float16* __restrict__ Vt, float* __restrict__ out)
{
    __shared__ __align__(16) _Float16 lds[8][16][64];   // 16 KB: per-wave P staging

    const int lane  = threadIdx.x & 31;
    const int wave  = threadIdx.x >> 5;
    const int w     = blockIdx.x * 8 + wave;
    const int bh    = w >> 7;
    const int t0    = (w & 127) << 4;
    const int b     = bh >> 4;
    const int h     = bh & 15;

    const int mrow  = lane & 15;
    const int abase = (lane < 16) ? 0 : 16;
    const int koff  = (lane < 16) ? 0 : 16;
    const int off   = (lane < 16) ? 0 : 8;
    const int ncol  = lane & 15;

    // Q A-operand (held for the whole kernel)
    v16h aq[2];
    {
        const _Float16* qrow = Qh + ((size_t)bh * T_SEQ + t0 + mrow) * DHEAD;
        #pragma unroll
        for (int c = 0; c < 2; ++c)
            aq[c] = loadA_h(qrow + c * 32, abase);
    }

    v8f o[4];
    #pragma unroll
    for (int nt = 0; nt < 4; ++nt)
        o[nt] = (v8f){0.f, 0.f, 0.f, 0.f, 0.f, 0.f, 0.f, 0.f};
    float mr[8], lr[8];
    #pragma unroll
    for (int r = 0; r < 8; ++r) { mr[r] = -1e30f; lr[r] = 0.f; }

    const _Float16* kbase = Kh + (size_t)bh * T_SEQ * DHEAD;
    const _Float16* vbase = Vt + (size_t)bh * DHEAD * T_SEQ;

    // prefetch first K/V blocks
    {
        // K block: 8 KB contiguous -> 256 B per lane (2 cachelines)
        const _Float16* kp = kbase + (size_t)lane * 128;
        __builtin_prefetch(kp, 0, 3);
        __builtin_prefetch(kp + 64, 0, 3);
        // V block: 64 segments of 128 B (stride T_SEQ halves) -> 2 segments/lane
        __builtin_prefetch(vbase + (size_t)(lane * 2) * T_SEQ, 0, 3);
        __builtin_prefetch(vbase + (size_t)(lane * 2 + 1) * T_SEQ, 0, 3);
    }

    for (int j = 0; j < T_SEQ / 64; ++j) {
        // ---- prefetch next 64-key block while this one computes ----
        if (j + 1 < T_SEQ / 64) {
            const _Float16* kp = kbase + (size_t)(j + 1) * 64 * DHEAD + (size_t)lane * 128;
            __builtin_prefetch(kp, 0, 3);
            __builtin_prefetch(kp + 64, 0, 3);
            const _Float16* vp = vbase + (size_t)(j + 1) * 64;
            __builtin_prefetch(vp + (size_t)(lane * 2) * T_SEQ, 0, 3);
            __builtin_prefetch(vp + (size_t)(lane * 2 + 1) * T_SEQ, 0, 3);
        }

        // ---- S = Q * K^T (16 x 64 block, log2-domain: Q pre-scaled) ----
        v8f s[4];
        #pragma unroll
        for (int nt = 0; nt < 4; ++nt) {
            v8f c0 = {0.f, 0.f, 0.f, 0.f, 0.f, 0.f, 0.f, 0.f};
            #pragma unroll
            for (int c = 0; c < 2; ++c) {
                v16h bk = load16h(kbase + (size_t)(j * 64 + nt * 16 + ncol) * DHEAD
                                        + c * 32 + koff);
                c0 = wmma16(aq[c], bk, c0);
            }
            s[nt] = c0;
        }

        // ---- online softmax (rows live across 16-lane halves, per VGPR r) ----
        float alpha[8];
        #pragma unroll
        for (int r = 0; r < 8; ++r) {
            float cm = fmaxf(fmaxf(s[0][r], s[1][r]), fmaxf(s[2][r], s[3][r]));
            #pragma unroll
            for (int msk = 8; msk >= 1; msk >>= 1)
                cm = fmaxf(cm, __shfl_xor(cm, msk, 32));
            const float nm = fmaxf(mr[r], cm);
            alpha[r] = __builtin_amdgcn_exp2f(mr[r] - nm);
            mr[r] = nm;
            float psum = 0.f;
            #pragma unroll
            for (int nt = 0; nt < 4; ++nt) {
                const float p = __builtin_amdgcn_exp2f(s[nt][r] - nm);
                s[nt][r] = p;
                psum += p;
            }
            lr[r] = lr[r] * alpha[r] + psum;
        }

        // ---- P: C/D layout -> LDS (row-major 16x64, private per wave) ----
        #pragma unroll
        for (int nt = 0; nt < 4; ++nt)
            #pragma unroll
            for (int r = 0; r < 8; ++r)
                lds[wave][r + off][nt * 16 + ncol] = (_Float16)s[nt][r];

        // rescale running O by alpha (per-row; uniform across each 16-lane half)
        #pragma unroll
        for (int nt = 0; nt < 4; ++nt)
            #pragma unroll
            for (int r = 0; r < 8; ++r)
                o[nt][r] *= alpha[r];

        // ---- reload P in A-operand layout (wave-private LDS; DS is in-order) ----
        v16h pa[2];
        #pragma unroll
        for (int c = 0; c < 2; ++c)
            pa[c] = loadA_h(&lds[wave][mrow][c * 32], abase);

        // ---- O += P * V  (Vt rows give contiguous B-operand loads) ----
        #pragma unroll
        for (int nt = 0; nt < 4; ++nt) {
            v8f c0 = o[nt];
            #pragma unroll
            for (int c = 0; c < 2; ++c) {
                v16h bv = load16h(vbase + (size_t)(nt * 16 + ncol) * T_SEQ
                                        + j * 64 + c * 32 + koff);
                c0 = wmma16(pa[c], bv, c0);
            }
            o[nt] = c0;
        }
    }

    // ---- finalize: divide by full row sum, write merged-head fp32 output ----
    #pragma unroll
    for (int r = 0; r < 8; ++r) {
        float l = lr[r];
        #pragma unroll
        for (int msk = 8; msk >= 1; msk >>= 1)
            l += __shfl_xor(l, msk, 32);
        const float inv = 1.0f / l;
        #pragma unroll
        for (int nt = 0; nt < 4; ++nt)
            out[((size_t)(b * T_SEQ + t0 + r + off)) * CDIM
                + h * DHEAD + nt * 16 + ncol] = o[nt][r] * inv;
    }
}

// ---------------------------------------------------------------------------
extern "C" void kernel_launch(void* const* d_in, const int* in_sizes, int n_in,
                              void* d_out, int out_size, void* d_ws, size_t ws_size,
                              hipStream_t stream) {
    (void)in_sizes; (void)n_in; (void)out_size; (void)ws_size;

    const float* x  = (const float*)d_in[0];
    const float* Wq = (const float*)d_in[1];
    const float* bq = (const float*)d_in[2];
    const float* Wk = (const float*)d_in[3];
    const float* bk = (const float*)d_in[4];
    const float* Wv = (const float*)d_in[5];
    const float* bv = (const float*)d_in[6];

    const size_t per = (size_t)BHTOT * T_SEQ * DHEAD;   // 8.39M halves each
    _Float16* Qh = (_Float16*)d_ws;
    _Float16* Kh = Qh + per;
    _Float16* Vt = Kh + per;

    // 8192 waves each: one wave per 16 rows of one (b,h)
    qkv_proj_kernel<<<1024, 256, 0, stream>>>(x, Wq, bq, Wk, bk, Wv, bv, Qh, Kh, Vt);
    flash_attn_kernel<<<1024, 256, 0, stream>>>(Qh, Kh, Vt, (float*)d_out);
}
